// Net_75874892252015
// MI455X (gfx1250) — compile-verified, matrix operand order checked by reading, not run on hardware
//
#include <hip/hip_runtime.h>
#include <hip/hip_bf16.h>

typedef __attribute__((ext_vector_type(16))) _Float16 v16h;
typedef __attribute__((ext_vector_type(8)))  float    v8f;

#define N_NODES 50000   // == 3125 * 16, so M-tiles divide exactly
#define N_EDGES 800000

// ---------------------------------------------------------------------------
// Degree / normalization
// ---------------------------------------------------------------------------
__global__ void init_deg_kernel(float* deg, int n) {
    int i = blockIdx.x * blockDim.x + threadIdx.x;
    if (i < n) deg[i] = 1.0f;  // self-loop
}

__global__ void count_deg_kernel(const long long* __restrict__ dst, float* deg, int e) {
    int i = blockIdx.x * blockDim.x + threadIdx.x;
    if (i < e) atomicAdd(&deg[(int)dst[i]], 1.0f);
}

__global__ void finalize_dinv_kernel(float* deg, int n) {
    int i = blockIdx.x * blockDim.x + threadIdx.x;
    if (i < n) deg[i] = rsqrtf(deg[i]);
}

// ---------------------------------------------------------------------------
// Pad x[N x 14] -> xpad[N x 32] (zero-filled) so layer-0 GEMM is guard-free.
// ---------------------------------------------------------------------------
__global__ void pad_x_kernel(const float* __restrict__ x, float* __restrict__ xp, int n) {
    int idx = blockIdx.x * blockDim.x + threadIdx.x;  // over n*32
    if (idx < n * 32) {
        int i = idx >> 5, k = idx & 31;
        xp[idx] = (k < 14) ? x[i * 14 + k] : 0.0f;
    }
}

// ---------------------------------------------------------------------------
// Pack W[Kreal x F] (f32) into WMMA B-fragment layout (f16), padded to Kpad.
// Fragment index: chunk c (Kpad/32), col-tile tn (F/16), lane (32), elem i (16).
// ISA 16-bit layout: lane holds column n = tn*16 + (lane&15); K sub-block by
// lane half: elems 0..7 -> k = c*32 + kh8 + i, elems 8..15 -> +16.
// ---------------------------------------------------------------------------
__global__ void pack_w_kernel(const float* __restrict__ W, _Float16* __restrict__ Wp,
                              int Kreal, int Kpad, int F) {
    int idx = blockIdx.x * blockDim.x + threadIdx.x;  // over Kpad*F halfs
    if (idx >= Kpad * F) return;
    int i     = idx & 15;
    int lane  = (idx >> 4) & 31;
    int t     = idx >> 9;
    int tilesF = F >> 4;
    int tn    = t % tilesF;
    int c     = t / tilesF;
    int kh8   = (lane >> 4) << 3;
    int k     = c * 32 + ((i < 8) ? (kh8 + i) : (16 + kh8 + (i - 8)));
    int n     = tn * 16 + (lane & 15);
    float v   = (k < Kreal) ? W[k * F + n] : 0.0f;
    Wp[idx] = (_Float16)v;
}

// ---------------------------------------------------------------------------
// WMMA GEMM: Out[N x F] = In[N x K] @ W[K x F], K % 32 == 0, N % 16 == 0.
// One wave per 16x16 output tile. A: 4x float4 contiguous loads + cvt per
// chunk; B: one 32-byte fragment load per lane from pre-packed Wp.
// Guard-free inner loop -> 6x b128 loads + cvts + 1 v_wmma per chunk.
// ---------------------------------------------------------------------------
__global__ void gemm_wmma_kernel(const float* __restrict__ In,
                                 const _Float16* __restrict__ Wp,
                                 float* __restrict__ Out,
                                 int N, int K, int F) {
    const int lane   = threadIdx.x & 31;
    const int wave   = blockIdx.x * (blockDim.x >> 5) + (threadIdx.x >> 5);
    const int tilesM = N >> 4;
    const int tilesF = F >> 4;
    if (wave >= tilesM * tilesF) return;  // wave-uniform exit (EXEC all-1s at WMMA)

    const int tm  = wave / tilesF;
    const int tn  = wave % tilesF;
    const int m   = tm * 16 + (lane & 15);
    const int kh8 = (lane >> 4) << 3;     // 0 or 8

    const float* arow = In + (long)m * K + kh8;
    const v16h*  bp   = (const v16h*)Wp + (long)tn * 32 + lane;
    const int    bstr = tilesF * 32;      // fragments per K-chunk

    v8f acc = {};
    const int nchunks = K >> 5;
    for (int c = 0; c < nchunks; ++c) {
        const float* ap = arow + (c << 5);
        float4 a0 = *(const float4*)(ap);
        float4 a1 = *(const float4*)(ap + 4);
        float4 a2 = *(const float4*)(ap + 16);
        float4 a3 = *(const float4*)(ap + 20);
        v16h a;
        a[0]  = (_Float16)a0.x; a[1]  = (_Float16)a0.y;
        a[2]  = (_Float16)a0.z; a[3]  = (_Float16)a0.w;
        a[4]  = (_Float16)a1.x; a[5]  = (_Float16)a1.y;
        a[6]  = (_Float16)a1.z; a[7]  = (_Float16)a1.w;
        a[8]  = (_Float16)a2.x; a[9]  = (_Float16)a2.y;
        a[10] = (_Float16)a2.z; a[11] = (_Float16)a2.w;
        a[12] = (_Float16)a3.x; a[13] = (_Float16)a3.y;
        a[14] = (_Float16)a3.z; a[15] = (_Float16)a3.w;
        v16h b = bp[(long)c * bstr];
        acc = __builtin_amdgcn_wmma_f32_16x16x32_f16(
            /*neg_a=*/false, a, /*neg_b=*/false, b,
            /*c_mod=*/(short)0, acc, /*reuse_a=*/false, /*reuse_b=*/false);
    }

    // C/D layout: elem r = D[tm*16 + 8*(lane>=16) + r][tn*16 + lane&15]
    const int mbase = tm * 16 + ((lane >> 4) << 3);
    const int n     = tn * 16 + (lane & 15);
#pragma unroll
    for (int r = 0; r < 8; ++r)
        Out[(long)(mbase + r) * F + n] = acc[r];
}

// ---------------------------------------------------------------------------
// Self-loop init: agg[i][f] = h[i][f] * dinv[i]^2
// ---------------------------------------------------------------------------
__global__ void init_agg_kernel(const float* __restrict__ h,
                                const float* __restrict__ dinv,
                                float* __restrict__ agg,
                                int n, int fshift) {
    long idx = (long)blockIdx.x * blockDim.x + threadIdx.x;
    long total = (long)n << fshift;
    if (idx < total) {
        int i = (int)(idx >> fshift);
        float di = dinv[i];
        agg[idx] = h[idx] * di * di;
    }
}

// ---------------------------------------------------------------------------
// Edge scatter: agg[dst] += h[src] * dinv[src] * dinv[dst]
// One thread = one edge x 4 features (float4 gather, 4 f32 atomics).
// Whole working set (h + agg + edges) is L2-resident on MI455X (192 MB).
// ---------------------------------------------------------------------------
__global__ void edge_scatter_kernel(const float* __restrict__ h,
                                    const long long* __restrict__ src,
                                    const long long* __restrict__ dst,
                                    const float* __restrict__ dinv,
                                    float* __restrict__ agg,
                                    int e, int fshift) {
    const int qshift = fshift - 2;            // float4 groups per row
    long idx = (long)blockIdx.x * blockDim.x + threadIdx.x;
    long total = (long)e << qshift;
    if (idx >= total) return;
    int ei = (int)(idx >> qshift);
    int g  = (int)(idx & ((1 << qshift) - 1));
    int s = (int)src[ei];
    int d = (int)dst[ei];
    float norm = dinv[s] * dinv[d];
    const float4 v = *(const float4*)(h + ((long)s << fshift) + 4 * g);
    float* o = agg + ((long)d << fshift) + 4 * g;
    atomicAdd(o + 0, v.x * norm);
    atomicAdd(o + 1, v.y * norm);
    atomicAdd(o + 2, v.z * norm);
    atomicAdd(o + 3, v.w * norm);
}

__global__ void relu_kernel(float* x, long n) {
    long i = (long)blockIdx.x * blockDim.x + threadIdx.x;
    if (i < n) x[i] = fmaxf(x[i], 0.0f);
}

// ---------------------------------------------------------------------------
// Sum pool over nodes: pooled[32] = sum_i h[i][0:32]
// ---------------------------------------------------------------------------
__global__ void zero_pooled_kernel(float* pooled) {
    if (threadIdx.x < 32) pooled[threadIdx.x] = 0.0f;
}

__global__ void pool_kernel(const float* __restrict__ h, float* pooled, int n) {
    // blockDim = 256: 8 row-lanes x 32 features per block
    int f         = threadIdx.x & 31;
    int rowStart  = blockIdx.x * 8 + (threadIdx.x >> 5);
    int rowStride = gridDim.x * 8;
    float s = 0.0f;
    for (int i = rowStart; i < n; i += rowStride)
        s += h[(long)i * 32 + f];
    atomicAdd(&pooled[f], s);
}

// ---------------------------------------------------------------------------
// Final MLP: out = relu(pooled @ fc11_w + b) @ fc12_w + b  (trivial size)
// ---------------------------------------------------------------------------
__global__ void mlp_kernel(const float* __restrict__ pooled,
                           const float* __restrict__ w1, const float* __restrict__ b1,
                           const float* __restrict__ w2, const float* __restrict__ b2,
                           float* out) {
    if (threadIdx.x == 0 && blockIdx.x == 0) {
        float a[16];
#pragma unroll
        for (int j = 0; j < 16; ++j) {
            float s = b1[j];
            for (int k = 0; k < 32; ++k) s += pooled[k] * w1[k * 16 + j];
            a[j] = fmaxf(s, 0.0f);
        }
        float o = b2[0];
#pragma unroll
        for (int j = 0; j < 16; ++j) o += a[j] * w2[j];
        out[0] = o;
    }
}

// ---------------------------------------------------------------------------
// Launcher
// ---------------------------------------------------------------------------
extern "C" void kernel_launch(void* const* d_in, const int* in_sizes, int n_in,
                              void* d_out, int out_size, void* d_ws, size_t ws_size,
                              hipStream_t stream) {
    const float*     x     = (const float*)d_in[0];
    const long long* edge  = (const long long*)d_in[1];  // int64 [2, E]
    const float*     W0    = (const float*)d_in[2];
    const float*     W1    = (const float*)d_in[3];
    const float*     W2    = (const float*)d_in[4];
    const float*     fc11w = (const float*)d_in[5];
    const float*     fc11b = (const float*)d_in[6];
    const float*     fc12w = (const float*)d_in[7];
    const float*     fc12b = (const float*)d_in[8];
    float*           out   = (float*)d_out;

    const int N = N_NODES;
    const int E = N_EDGES;
    const long long* src = edge;
    const long long* dst = edge + E;

    // Workspace layout (all offsets 32B-aligned):
    // dinv[N] | bufA[N*128] | bufB[N*128] | xpad[N*32] | pooled[32] |
    // Wp0[32*128 f16] | Wp1[128*128 f16] | Wp2[128*32 f16]
    float*     dinv   = (float*)d_ws;
    float*     bufA   = dinv + N;
    float*     bufB   = bufA + (long)N * 128;
    float*     xpad   = bufB + (long)N * 128;
    float*     pooled = xpad + (long)N * 32;
    _Float16*  Wp0    = (_Float16*)(pooled + 32);
    _Float16*  Wp1    = Wp0 + 32 * 128;
    _Float16*  Wp2    = Wp1 + 128 * 128;

    const int TB = 256;
    auto grid1 = [](long n, int tb) { return (int)((n + tb - 1) / tb); };

    // --- degrees -> d_inv ---
    init_deg_kernel<<<grid1(N, TB), TB, 0, stream>>>(dinv, N);
    count_deg_kernel<<<grid1(E, TB), TB, 0, stream>>>(dst, dinv, E);
    finalize_dinv_kernel<<<grid1(N, TB), TB, 0, stream>>>(dinv, N);

    // --- prep: pad x, pack weights into f16 B-fragments ---
    pad_x_kernel<<<grid1((long)N * 32, TB), TB, 0, stream>>>(x, xpad, N);
    pack_w_kernel<<<grid1(32 * 128, TB), TB, 0, stream>>>(W0, Wp0, 14, 32, 128);
    pack_w_kernel<<<grid1(128 * 128, TB), TB, 0, stream>>>(W1, Wp1, 128, 128, 128);
    pack_w_kernel<<<grid1(128 * 32, TB), TB, 0, stream>>>(W2, Wp2, 128, 128, 32);

    // --- layer 0: 14(->32) -> 128 ---
    {
        int tiles = (N / 16) * (128 / 16);
        gemm_wmma_kernel<<<(tiles + 7) / 8, TB, 0, stream>>>(xpad, Wp0, bufB, N, 32, 128);
        init_agg_kernel<<<grid1((long)N * 128, TB), TB, 0, stream>>>(bufB, dinv, bufA, N, 7);
        edge_scatter_kernel<<<grid1((long)E * 32, TB), TB, 0, stream>>>(bufB, src, dst, dinv, bufA, E, 7);
        relu_kernel<<<grid1((long)N * 128, TB), TB, 0, stream>>>(bufA, (long)N * 128);
    }
    // --- layer 1: 128 -> 128 ---
    {
        int tiles = (N / 16) * (128 / 16);
        gemm_wmma_kernel<<<(tiles + 7) / 8, TB, 0, stream>>>(bufA, Wp1, bufB, N, 128, 128);
        init_agg_kernel<<<grid1((long)N * 128, TB), TB, 0, stream>>>(bufB, dinv, bufA, N, 7);
        edge_scatter_kernel<<<grid1((long)E * 32, TB), TB, 0, stream>>>(bufB, src, dst, dinv, bufA, E, 7);
        relu_kernel<<<grid1((long)N * 128, TB), TB, 0, stream>>>(bufA, (long)N * 128);
    }
    // --- layer 2: 128 -> 32 ---
    {
        int tiles = (N / 16) * (32 / 16);
        gemm_wmma_kernel<<<(tiles + 7) / 8, TB, 0, stream>>>(bufA, Wp2, bufB, N, 128, 32);
        init_agg_kernel<<<grid1((long)N * 32, TB), TB, 0, stream>>>(bufB, dinv, bufA, N, 5);
        edge_scatter_kernel<<<grid1((long)E * 8, TB), TB, 0, stream>>>(bufB, src, dst, dinv, bufA, E, 5);
        relu_kernel<<<grid1((long)N * 32, TB), TB, 0, stream>>>(bufA, (long)N * 32);
    }

    // --- pool + MLP ---
    zero_pooled_kernel<<<1, 32, 0, stream>>>(pooled);
    pool_kernel<<<256, TB, 0, stream>>>(bufA, pooled, N);
    mlp_kernel<<<1, 32, 0, stream>>>(pooled, fc11w, fc11b, fc12w, fc12b, out);
}
